// BahdanauAttnDecoderRNN_77532749627886
// MI455X (gfx1250) — matrix operand order repaired and checked by reading, not compiled, not run on hardware
//
#include <hip/hip_runtime.h>
#include <hip/hip_bf16.h>

// ---------------------------------------------------------------------------
// BahdanauAttnDecoderRNN single step, bf16-WMMA path for gfx1250 (MI455X).
// B=64, S=400, E=512, H=1024, A=512, V=50257.
// Round 2: LDS staging of the A operand (ds_load_b128 fragment reads),
// eliminating the 4x-32x redundant global A traffic; weight-stream prefetch.
// ---------------------------------------------------------------------------

typedef __bf16 bf16_t;
typedef __attribute__((ext_vector_type(16))) __bf16 v16bf;
typedef __attribute__((ext_vector_type(8)))  __bf16 v8bf;
typedef __attribute__((ext_vector_type(4)))  __bf16 v4bf;
typedef __attribute__((ext_vector_type(8)))  float  v8f;

#define WMMA_BF16(a, b, c) \
  __builtin_amdgcn_wmma_f32_16x16x32_bf16(false, (a), false, (b), (short)0, (c), false, false)

static __device__ inline v8f v8f_zero() {
  v8f z;
#pragma unroll
  for (int i = 0; i < 8; ++i) z[i] = 0.0f;
  return z;
}

// A-fragment (16x32 bf16): lane&15 = row; lanes>=16 shift K base by 16.
// per-lane K values: {kb..kb+15} -> elems 0..15.
static __device__ inline v16bf load_a_bf16(const bf16_t* A, int row, int kb, int lda) {
  const bf16_t* p = A + (size_t)row * lda + kb;
  v16bf r;
#pragma unroll
  for (int i = 0; i < 16; ++i) r[i] = p[i];
  return r;
}

// B-fragment (32x16 bf16) from row-major weight W[N,K]: lane&15 = column n,
// lanes>=16 take K base +16; 16 contiguous K per lane, converted f32->bf16.
static __device__ inline v16bf load_b_f32(const float* __restrict__ W, int row, int kb, int ldw) {
  const float* p = W + (size_t)row * ldw + kb;
  v16bf r;
#pragma unroll
  for (int i = 0; i < 16; ++i) r[i] = (bf16_t)p[i];
  return r;
}

// ---------------------------------------------------------------------------
// K1: build A0 = [emb(512) | input_feed(1024) | h0(1024)] as bf16 [64 x 2560]
// ---------------------------------------------------------------------------
__global__ void __launch_bounds__(256) prep_kernel(
    const int* __restrict__ tokens, const float* __restrict__ emb_table,
    const float* __restrict__ input_feed, const float* __restrict__ h0,
    bf16_t* __restrict__ A0) {
  int idx = blockIdx.x * 256 + threadIdx.x;
  if (idx >= 64 * 2560) return;
  int b = idx / 2560, j = idx - b * 2560;
  float v;
  if (j < 512)       v = emb_table[(size_t)tokens[b] * 512 + j];
  else if (j < 1536) v = input_feed[b * 1024 + (j - 512)];
  else               v = h0[b * 1024 + (j - 1536)];
  A0[idx] = (bf16_t)v;
}

// ---------------------------------------------------------------------------
// K2: gates[64,4096] = A0 @ [W_ih | W_hh]^T + b_ih + b_hh   (K = 2560)
// A staged slab-wise (64x512 bf16 = 64 KB) in LDS; W switch at k=1536 falls
// on a slab boundary.
// ---------------------------------------------------------------------------
__global__ void __launch_bounds__(128) gates_kernel(
    const bf16_t* __restrict__ A0, const float* __restrict__ W_ih,
    const float* __restrict__ W_hh, const float* __restrict__ b_ih,
    const float* __restrict__ b_hh, float* __restrict__ gates) {
  __shared__ bf16_t As[64 * 512];
  const int lane = threadIdx.x & 31;
  const int wave = threadIdx.x >> 5;
  const int n0   = blockIdx.x * 64 + wave * 16;
  const int col  = n0 + (lane & 15);
  const int kbB  = (lane >> 4) << 4;
  const int kbA  = (lane >> 4) << 4;
  const int rowA = lane & 15;
  v8f acc[4];
#pragma unroll
  for (int t = 0; t < 4; ++t) acc[t] = v8f_zero();

  for (int ks = 0; ks < 2560; ks += 512) {
    __syncthreads();
    for (int i = threadIdx.x * 8; i < 64 * 512; i += 128 * 8) {
      int row = i >> 9, off = i & 511;
      *(v8bf*)(As + i) = *(const v8bf*)(A0 + (size_t)row * 2560 + ks + off);
    }
    __syncthreads();
    const float* Wp; int ldw, kk;
    if (ks < 1536) { Wp = W_ih; ldw = 1536; kk = ks; }
    else           { Wp = W_hh; ldw = 1024; kk = ks - 1536; }
    for (int k0 = 0; k0 < 512; k0 += 32) {
      v16bf bf = load_b_f32(Wp, col, kk + k0 + kbB, ldw);
      __builtin_prefetch(Wp + (size_t)col * ldw + kk + k0 + kbB + 128, 0, 1);
#pragma unroll
      for (int t = 0; t < 4; ++t) {
        v16bf a = load_a_bf16(As, 16 * t + rowA, k0 + kbA, 512);
        acc[t] = WMMA_BF16(a, bf, acc[t]);
      }
    }
  }
  const float bv = b_ih[col] + b_hh[col];
  const int rbase = (lane >> 4) * 8;
#pragma unroll
  for (int t = 0; t < 4; ++t)
#pragma unroll
    for (int r = 0; r < 8; ++r)
      gates[(size_t)(16 * t + rbase + r) * 4096 + col] = acc[t][r] + bv;
}

// ---------------------------------------------------------------------------
// K3: LSTM elementwise. Writes c1, h1 (fp32) and h1 as bf16 into cat[:,0:1024]
// ---------------------------------------------------------------------------
__global__ void __launch_bounds__(256) lstm_kernel(
    const float* __restrict__ gates, const float* __restrict__ c0,
    float* __restrict__ h1_out, float* __restrict__ c1_out,
    bf16_t* __restrict__ cat) {
  int idx = blockIdx.x * 256 + threadIdx.x;
  if (idx >= 64 * 1024) return;
  int b = idx >> 10, h = idx & 1023;
  const float* g = gates + (size_t)b * 4096;
  float gi = g[h], gf = g[1024 + h], gg = g[2048 + h], go = g[3072 + h];
  float si = 1.0f / (1.0f + __expf(-gi));
  float sf = 1.0f / (1.0f + __expf(-gf));
  float so = 1.0f / (1.0f + __expf(-go));
  float c1 = sf * c0[idx] + si * tanhf(gg);
  float h1 = so * tanhf(c1);
  c1_out[idx] = c1;
  h1_out[idx] = h1;
  cat[(size_t)b * 2048 + h] = (bf16_t)h1;
}

// ---------------------------------------------------------------------------
// Generic M=64 GEMM: C[64,N] = A(bf16)[64,K] @ W(f32)[N,K]^T + bias.
// A staged slab-wise in LDS; optional bf16 mirror of C. K % 512 == 0.
// ---------------------------------------------------------------------------
__global__ void __launch_bounds__(128) gemm_a64_kernel(
    const bf16_t* __restrict__ A, int lda, int K,
    const float* __restrict__ W, const float* __restrict__ bias,
    float* __restrict__ C, int ldc, int N,
    bf16_t* __restrict__ Cbf, int ldcb) {
  __shared__ bf16_t As[64 * 512];
  const int lane = threadIdx.x & 31;
  const int wave = threadIdx.x >> 5;
  const int n0   = blockIdx.x * 64 + wave * 16;
  const int col  = n0 + (lane & 15);
  const int wrow = (col < N) ? col : (N - 1);   // clamp OOB weight rows
  const int kbB  = (lane >> 4) << 4;
  const int kbA  = (lane >> 4) << 4;
  const int rowA = lane & 15;
  v8f acc[4];
#pragma unroll
  for (int t = 0; t < 4; ++t) acc[t] = v8f_zero();

  for (int ks = 0; ks < K; ks += 512) {
    __syncthreads();
    for (int i = threadIdx.x * 8; i < 64 * 512; i += 128 * 8) {
      int row = i >> 9, off = i & 511;
      *(v8bf*)(As + i) = *(const v8bf*)(A + (size_t)row * lda + ks + off);
    }
    __syncthreads();
    for (int k0 = 0; k0 < 512; k0 += 32) {
      v16bf bf = load_b_f32(W, wrow, ks + k0 + kbB, K);
      __builtin_prefetch(W + (size_t)wrow * K + ks + k0 + kbB + 128, 0, 1);
#pragma unroll
      for (int t = 0; t < 4; ++t) {
        v16bf a = load_a_bf16(As, 16 * t + rowA, k0 + kbA, 512);
        acc[t] = WMMA_BF16(a, bf, acc[t]);
      }
    }
  }
  if (col < N) {
    const float bv = bias[col];
    const int rbase = (lane >> 4) * 8;
#pragma unroll
    for (int t = 0; t < 4; ++t)
#pragma unroll
      for (int r = 0; r < 8; ++r) {
        int row = 16 * t + rbase + r;
        float v = acc[t][r] + bv;
        C[(size_t)row * ldc + col] = v;
        if (Cbf) Cbf[(size_t)row * ldcb + col] = (bf16_t)v;
      }
  }
}

// ---------------------------------------------------------------------------
// K5: alignment[s,b] = align_w . tanh(query[b,:] + attnm_w @ enc[s,b,:])
// One block per s. Encoder tile (64x1024) staged ONCE into LDS as bf16
// (128 KB of the 320 KB WGP LDS); keys GEMM via WMMA with ds_load fragments;
// fused tanh + align_w reduction through LDS float atomics.
// ---------------------------------------------------------------------------
__global__ void __launch_bounds__(128) attn_score_kernel(
    const float* __restrict__ enc,      // [S,B,H]
    const float* __restrict__ attnm_w,  // [512,1024]
    const float* __restrict__ query,    // [64,512] (incl. attnq_b)
    const float* __restrict__ align_w,  // [512]
    float* __restrict__ alignment) {    // [S*64]
  extern __shared__ char smem[];
  bf16_t* encs = (bf16_t*)smem;                 // 64*1024 bf16 = 128 KB
  float*  part = (float*)(smem + 64 * 1024 * 2);  // 64 floats
  const int s = blockIdx.x;
  const float* encS = enc + (size_t)s * 64 * 1024;

  // stage encoder tile, converting f32 -> bf16 (coalesced float4 reads)
  for (int i = threadIdx.x * 4; i < 64 * 1024; i += 128 * 4) {
    float4 v = *(const float4*)(encS + i);
    v4bf b;
    b[0] = (bf16_t)v.x; b[1] = (bf16_t)v.y;
    b[2] = (bf16_t)v.z; b[3] = (bf16_t)v.w;
    *(v4bf*)(encs + i) = b;
  }
  if (threadIdx.x < 64) part[threadIdx.x] = 0.0f;
  __syncthreads();

  const int lane = threadIdx.x & 31;
  const int wave = threadIdx.x >> 5;
  const int kbB  = (lane >> 4) << 4;
  const int kbA  = (lane >> 4) << 4;
  const int rowA = lane & 15;
  const int rbase = (lane >> 4) * 8;

  for (int chunk = 0; chunk < 8; ++chunk) {
    const int n0  = chunk * 64 + wave * 16;
    const int col = n0 + (lane & 15);
    v8f acc[4];
#pragma unroll
    for (int t = 0; t < 4; ++t) acc[t] = v8f_zero();
    for (int k0 = 0; k0 < 1024; k0 += 32) {
      v16bf bf = load_b_f32(attnm_w, col, k0 + kbB, 1024);
#pragma unroll
      for (int t = 0; t < 4; ++t) {
        v16bf a = load_a_bf16(encs, 16 * t + rowA, k0 + kbA, 1024);
        acc[t] = WMMA_BF16(a, bf, acc[t]);
      }
    }
    const float aw = align_w[col];
#pragma unroll
    for (int t = 0; t < 4; ++t)
#pragma unroll
      for (int r = 0; r < 8; ++r) {
        int row = 16 * t + rbase + r;  // = batch index b
        float v = tanhf(query[row * 512 + col] + acc[t][r]) * aw;
        atomicAdd(&part[row], v);
      }
  }
  __syncthreads();
  if (threadIdx.x < 64) alignment[s * 64 + threadIdx.x] = part[threadIdx.x];
}

// ---------------------------------------------------------------------------
// K6: softmax over S per batch + context ctx[b,:] = sum_s w[s,b]*enc[s,b,:].
// One block per b. Writes attn weights (fp32) and ctx (bf16 into cat[:,1024:]).
// ---------------------------------------------------------------------------
__global__ void __launch_bounds__(256) softmax_ctx_kernel(
    const float* __restrict__ alignment, const float* __restrict__ enc,
    float* __restrict__ attn_out, bf16_t* __restrict__ cat) {
  __shared__ float wsm[400];
  __shared__ float red[256];
  const int b = blockIdx.x, t = threadIdx.x;

  float m = -3.4e38f;
  for (int s = t; s < 400; s += 256) m = fmaxf(m, alignment[s * 64 + b]);
  red[t] = m; __syncthreads();
  for (int off = 128; off > 0; off >>= 1) {
    if (t < off) red[t] = fmaxf(red[t], red[t + off]);
    __syncthreads();
  }
  const float mx = red[0];
  __syncthreads();

  float sum = 0.0f;
  for (int s = t; s < 400; s += 256) {
    float e = __expf(alignment[s * 64 + b] - mx);
    wsm[s] = e; sum += e;
  }
  red[t] = sum; __syncthreads();
  for (int off = 128; off > 0; off >>= 1) {
    if (t < off) red[t] += red[t + off];
    __syncthreads();
  }
  const float inv = 1.0f / red[0];
  for (int s = t; s < 400; s += 256) {
    float v = wsm[s] * inv;
    wsm[s] = v;
    attn_out[s * 64 + b] = v;
  }
  __syncthreads();

  float a0 = 0, a1 = 0, a2 = 0, a3 = 0;
  const float* eb = enc + (size_t)b * 1024;
  for (int s = 0; s < 400; ++s) {
    float w = wsm[s];
    const float* p = eb + (size_t)s * 65536;  // 64*1024
    a0 += w * p[t];       a1 += w * p[t + 256];
    a2 += w * p[t + 512]; a3 += w * p[t + 768];
  }
  bf16_t* cb = cat + (size_t)b * 2048 + 1024;
  cb[t] = (bf16_t)a0; cb[t + 256] = (bf16_t)a1;
  cb[t + 512] = (bf16_t)a2; cb[t + 768] = (bf16_t)a3;
}

// ---------------------------------------------------------------------------
extern "C" void kernel_launch(void* const* d_in, const int* in_sizes, int n_in,
                              void* d_out, int out_size, void* d_ws, size_t ws_size,
                              hipStream_t stream) {
  (void)in_sizes; (void)n_in; (void)out_size; (void)ws_size;
  const int*   tokens  = (const int*)d_in[0];
  const float* h0      = (const float*)d_in[1];
  const float* c0      = (const float*)d_in[2];
  const float* enc     = (const float*)d_in[3];
  const float* infeed  = (const float*)d_in[4];
  const float* embt    = (const float*)d_in[5];
  const float* W_ih    = (const float*)d_in[6];
  const float* W_hh    = (const float*)d_in[7];
  const float* b_ih    = (const float*)d_in[8];
  const float* b_hh    = (const float*)d_in[9];
  const float* attnm_w = (const float*)d_in[10];
  const float* attnq_w = (const float*)d_in[11];
  const float* attnq_b = (const float*)d_in[12];
  const float* align_w = (const float*)d_in[13];
  const float* lin_w   = (const float*)d_in[14];
  const float* lin_b   = (const float*)d_in[15];
  const float* out_w   = (const float*)d_in[16];
  const float* out_b   = (const float*)d_in[17];

  // d_out layout: logits | h1 | c1 | attn_weights | input_feed_new
  float* out    = (float*)d_out;
  float* logits = out;
  float* h1o    = out + (size_t)64 * 50257;
  float* c1o    = h1o + 64 * 1024;
  float* attno  = c1o + 64 * 1024;
  float* feedo  = attno + 400 * 64;

  // workspace layout (≈2.0 MB)
  char*   ws    = (char*)d_ws;
  bf16_t* A0    = (bf16_t*)(ws);            // [64,2560] bf16   (327680 B)
  float*  gates = (float*)(ws + 327680);    // [64,4096] f32    (1048576 B)
  bf16_t* cat   = (bf16_t*)(ws + 1376256);  // [64,2048] bf16   (262144 B)
  float*  query = (float*)(ws + 1638400);   // [64,512]  f32    (131072 B)
  float*  alg   = (float*)(ws + 1769472);   // [400,64]  f32    (102400 B)
  bf16_t* feedb = (bf16_t*)(ws + 1871872);  // [64,1024] bf16   (131072 B)

  hipLaunchKernelGGL(prep_kernel, dim3(640), dim3(256), 0, stream,
                     tokens, embt, infeed, h0, A0);
  hipLaunchKernelGGL(gates_kernel, dim3(64), dim3(128), 0, stream,
                     A0, W_ih, W_hh, b_ih, b_hh, gates);
  hipLaunchKernelGGL(lstm_kernel, dim3(256), dim3(256), 0, stream,
                     gates, c0, h1o, c1o, cat);
  // query = h1 @ attnq_w^T + attnq_b   (A = cat cols 0..1023, lda=2048)
  hipLaunchKernelGGL(gemm_a64_kernel, dim3(8), dim3(128), 0, stream,
                     cat, 2048, 1024, attnq_w, attnq_b, query, 512, 512,
                     (bf16_t*)nullptr, 0);
  // keys + alignment, 128 KB + 256 B dynamic LDS
  hipLaunchKernelGGL(attn_score_kernel, dim3(400), dim3(128),
                     64 * 1024 * 2 + 64 * 4, stream,
                     enc, attnm_w, query, align_w, alg);
  hipLaunchKernelGGL(softmax_ctx_kernel, dim3(64), dim3(256), 0, stream,
                     alg, enc, attno, cat);
  // input_feed_new = cat @ lin_out_w^T + b   (K=2048)
  hipLaunchKernelGGL(gemm_a64_kernel, dim3(16), dim3(128), 0, stream,
                     cat, 2048, 2048, lin_w, lin_b, feedo, 1024, 1024,
                     feedb, 1024);
  // logits = feed @ out_w^T + out_b   (N=50257, guarded)
  hipLaunchKernelGGL(gemm_a64_kernel, dim3(786), dim3(128), 0, stream,
                     feedb, 1024, 1024, out_w, out_b, logits, 50257, 50257,
                     (bf16_t*)nullptr, 0);
}
